// ParticleNet_51204600103278
// MI455X (gfx1250) — compile-verified, matrix operand order checked by reading, not run on hardware
//
#include <hip/hip_runtime.h>
#include <hip/hip_bf16.h>

// ---------------------------------------------------------------------------
// ParticleNet-style DGCNN forward for MI455X (gfx1250, wave32, WMMA bf16).
//
// Fixed problem: B=512 graphs, P=128 nodes, D=4, k=16, conv=(64,128,256),
// fc=(256,128), out=2.  Dominant cost: EdgeConv MLP GEMMs with
// M = B*P*k = 1,048,576 rows -> mapped to v_wmma_f32_16x16x32_bf16.
//
// Training-mode BatchNorm => biases before BN cancel; BN needs column
// mean/var over all M rows, done with a stats pass + recompute pass
// (features stay resident in the 192MB L2, so recompute is cheap vs.
// storing ~1GB of pre-BN activations).
// ---------------------------------------------------------------------------

#define EPSV 1e-5f

typedef __attribute__((ext_vector_type(16))) __bf16 v16bf;
typedef __attribute__((ext_vector_type(8)))  float  v8f;

union FragB16 { v16bf v; uint4 q[2]; };

__device__ inline unsigned short f2bf(float f) {
  unsigned u = __builtin_bit_cast(unsigned, f);
  unsigned r = u + 0x7FFFu + ((u >> 16) & 1u);
  return (unsigned short)(r >> 16);
}

// Load a 16x32 bf16 fragment (A or B^T) stored contiguously along K.
// Per ISA 7.12.2 (16-bit A 16x32): lane L holds row L%16; lanes 0-15 get
// K = k0..k0+7 and k0+16..k0+23, lanes 16-31 get K = k0+8..15, k0+24..31.
// Caller passes p0 = base + row*stride + k0 + ((lane>>4)*8).
__device__ inline v16bf load_frag16(const unsigned short* p0) {
  FragB16 f;
  f.q[0] = *(const uint4*)(p0);
  f.q[1] = *(const uint4*)(p0 + 16);
  return f.v;
}

__device__ inline v8f wmma_bf16(v16bf a, v16bf b, v8f c) {
  return __builtin_amdgcn_wmma_f32_16x16x32_bf16(
      /*neg_a=*/false, a, /*neg_b=*/false, b,
      /*c_mod=*/(short)0, c, /*reuse_a=*/false, /*reuse_b=*/false);
}

// ---------------------------------------------------------------------------
// Utility kernels
// ---------------------------------------------------------------------------

__global__ __launch_bounds__(256) void zero_kernel(float* __restrict__ p, int n) {
  int t = blockIdx.x * 256 + threadIdx.x;
  if (t < n) p[t] = 0.f;
}

// Column sum / sum-of-squares of X[M, C] (f32) -> S[C], Q[C] (atomic accumulate).
__global__ __launch_bounds__(256) void colstats_kernel(
    const float* __restrict__ X, int M, int C,
    float* __restrict__ S, float* __restrict__ Q) {
  extern __shared__ char smem[];
  float* cs = (float*)smem;  // 2*C floats
  for (int t = threadIdx.x; t < 2 * C; t += 256) cs[t] = 0.f;
  __syncthreads();
  long long total = (long long)M * C;
  for (long long t = (long long)blockIdx.x * 256 + threadIdx.x; t < total;
       t += (long long)gridDim.x * 256) {
    int c = (int)(t % C);
    float v = X[t];
    atomicAdd(&cs[c], v);
    atomicAdd(&cs[C + c], v * v);
  }
  __syncthreads();
  for (int t = threadIdx.x; t < C; t += 256) {
    atomicAdd(&S[t], cs[t]);
    atomicAdd(&Q[t], cs[C + t]);
  }
}

// y = g*(x-m)*rsqrt(v+eps)+b   (input BN, no ReLU)
__global__ __launch_bounds__(256) void bn_input_kernel(
    const float* __restrict__ x, const float* __restrict__ S,
    const float* __restrict__ Q, const float* __restrict__ g,
    const float* __restrict__ b, float* __restrict__ y, int M, int C) {
  int t = blockIdx.x * 256 + threadIdx.x;
  if (t >= M * C) return;
  int c = t % C;
  float invM = 1.f / (float)M;
  float m = S[c] * invM;
  float var = Q[c] * invM - m * m;
  float sc = g[c] * rsqrtf(var + EPSV);
  y[t] = sc * (x[t] - m) + b[c];
}

// In-place BN + ReLU on Z[M, C]
__global__ __launch_bounds__(256) void bn_relu_kernel(
    float* __restrict__ Z, const float* __restrict__ S,
    const float* __restrict__ Q, const float* __restrict__ g,
    const float* __restrict__ b, int M, int C) {
  int t = blockIdx.x * 256 + threadIdx.x;
  if (t >= M * C) return;
  int c = t % C;
  float invM = 1.f / (float)M;
  float m = S[c] * invM;
  float var = Q[c] * invM - m * m;
  float sc = g[c] * rsqrtf(var + EPSV);
  Z[t] = fmaxf(sc * (Z[t] - m) + b[c], 0.f);
}

// Convert W[K, N] (f32, row-major) -> Wt[N, KP] bf16 (transposed, K-padded with 0)
__global__ __launch_bounds__(256) void convw_kernel(
    const float* __restrict__ W, unsigned short* __restrict__ Wt,
    int K, int N, int KP) {
  int t = blockIdx.x * 256 + threadIdx.x;
  if (t >= N * KP) return;
  int n = t / KP, k = t - n * KP;
  Wt[t] = (k < K) ? f2bf(W[k * N + n]) : (unsigned short)0;
}

// Small dense GEMM (FC tail): Z[M,N] = H[M,K] @ W[K,N]  (bias cancels in BN)
__global__ __launch_bounds__(256) void fc_gemm_kernel(
    const float* __restrict__ H, const float* __restrict__ W,
    float* __restrict__ Z, int M, int K, int N) {
  int t = blockIdx.x * 256 + threadIdx.x;
  if (t >= M * N) return;
  int m = t / N, n = t - m * N;
  float a = 0.f;
  for (int k = 0; k < K; ++k) a += H[m * K + k] * W[k * N + n];
  Z[t] = a;
}

// Final linear with bias: out[M,N] = H[M,K] @ W[K,N] + b
__global__ __launch_bounds__(256) void out_kernel(
    const float* __restrict__ H, const float* __restrict__ W,
    const float* __restrict__ b, float* __restrict__ out, int M, int K, int N) {
  int t = blockIdx.x * 256 + threadIdx.x;
  if (t >= M * N) return;
  int m = t / N, n = t - m * N;
  float a = b[n];
  for (int k = 0; k < K; ++k) a += H[m * K + k] * W[k * N + n];
  out[t] = a;
}

// Global mean pool over P=128 nodes per graph: pooled[g, off+c] = mean_p x[g,p,c]
template <int COUT>
__global__ __launch_bounds__(256) void pool_kernel(
    const float* __restrict__ xn, float* __restrict__ pooled, int off) {
  int g = blockIdx.x;
  for (int c = threadIdx.x; c < COUT; c += 256) {
    float s = 0.f;
    for (int p = 0; p < 128; ++p) s += xn[(g * 128 + p) * COUT + c];
    pooled[g * 448 + off + c] = s * (1.f / 128.f);
  }
}

// ---------------------------------------------------------------------------
// kNN (k=16) on first 2 feature dims, per graph (P=128). One block per graph.
// ---------------------------------------------------------------------------
template <int CIN>
__global__ __launch_bounds__(128) void knn_kernel(
    const float* __restrict__ x, int* __restrict__ idxb) {
  __shared__ float px[128], py[128];
  const int g = blockIdx.x, t = threadIdx.x;
  const int node = g * 128 + t;
  px[t] = x[node * CIN + 0];
  py[t] = x[node * CIN + 1];
  __syncthreads();
  float bd[16];
  int bi[16];
#pragma unroll
  for (int q = 0; q < 16; ++q) { bd[q] = 1e30f; bi[q] = 0; }
  const float xx = px[t], yy = py[t];
  for (int m = 0; m < 128; ++m) {
    if (m == t) continue;
    float dx = xx - px[m], dy = yy - py[m];
    float d = dx * dx + dy * dy;
    if (d < bd[15]) {
      int pos = 15;
      for (int q = 14; q >= 0; --q) {
        if (d < bd[q]) { bd[q + 1] = bd[q]; bi[q + 1] = bi[q]; pos = q; }
        else break;
      }
      bd[pos] = d;
      bi[pos] = m;
    }
  }
  for (int q = 0; q < 16; ++q) idxb[node * 16 + q] = bi[q];
}

// ---------------------------------------------------------------------------
// EdgeConv WMMA kernels.  M-tile = MT edge rows per workgroup (256 thr, 8 waves).
// Wave w owns row-group (w % RG)*16 (one node's 16 edges) and an nt-subset.
// ---------------------------------------------------------------------------

template <int CIN, int COUT, int MT>
struct EC {
  static constexpr int K2 = 2 * CIN;
  static constexpr int K2P = (K2 < 32) ? 32 : K2;
  static constexpr int RG = MT / 16;   // row groups per WG
  static constexpr int NS = 8 / RG;    // waves sharing a row group (nt split)
  static constexpr int NT = COUT / 16;
  static constexpr int NTW = NT / NS;  // nt tiles per wave
  static constexpr size_t SM1 =
      (size_t)MT * K2P * 2 + (size_t)2 * COUT * 4 + (size_t)MT * 4;
  static constexpr size_t SM23 = (size_t)MT * K2P * 2 + (size_t)MT * COUT * 2 +
                                 (size_t)4 * COUT * 4 + (size_t)MT * 4;
};

// Build the MT x K2P bf16 edge-feature tile [xi | xj - xi] in LDS.
template <int CIN, int COUT, int MT>
__device__ inline void build_Atile(const float* __restrict__ x,
                                   const int* __restrict__ idxb,
                                   unsigned short* As, int* jrow, int e0,
                                   int tid) {
  using E = EC<CIN, COUT, MT>;
  for (int r = tid; r < MT; r += 256) {
    int e = e0 + r;
    int i = e >> 4;
    jrow[r] = (i & ~127) + idxb[e];  // graph-local -> global node id
  }
  __syncthreads();
  for (int t = tid; t < MT * E::K2P; t += 256) {
    int r = t / E::K2P;
    int kk = t - r * E::K2P;
    int i = (e0 + r) >> 4;
    float a;
    if (kk < CIN) a = x[i * CIN + kk];
    else if (kk < E::K2) {
      int c = kk - CIN;
      a = x[jrow[r] * CIN + c] - x[i * CIN + c];
    } else a = 0.f;
    As[r * E::K2P + kk] = f2bf(a);
  }
  __syncthreads();
}

// Pass 1: z1 = edge_feats @ W1 ; accumulate per-column sum / sumsq of z1.
template <int CIN, int COUT, int MT>
__global__ __launch_bounds__(256) void edge_gemm1_stats(
    const float* __restrict__ x, const int* __restrict__ idxb,
    const unsigned short* __restrict__ Wt1, float* __restrict__ gsum,
    float* __restrict__ gsq) {
  using E = EC<CIN, COUT, MT>;
  extern __shared__ char smem[];
  unsigned short* As = (unsigned short*)smem;
  float* cs = (float*)(smem + (size_t)MT * E::K2P * 2);
  float* cq = cs + COUT;
  int* jrow = (int*)(cq + COUT);
  const int tid = threadIdx.x;
  const int e0 = blockIdx.x * MT;

  for (int t = tid; t < 2 * COUT; t += 256) cs[t] = 0.f;
  build_Atile<CIN, COUT, MT>(x, idxb, As, jrow, e0, tid);

  const int wid = tid >> 5, lane = tid & 31;
  const int rg = wid % E::RG, nt0 = wid / E::RG;
  const int rowb = rg * 16 + (lane & 15);
  const int kb = (lane >> 4) * 8;
  const int coll = lane & 15;

  for (int ni = 0; ni < E::NTW; ++ni) {
    const int nt = nt0 + ni * E::NS;
    v8f acc = {0.f, 0.f, 0.f, 0.f, 0.f, 0.f, 0.f, 0.f};
#pragma unroll
    for (int kt = 0; kt < E::K2P / 32; ++kt) {
      v16bf a = load_frag16(As + rowb * E::K2P + kt * 32 + kb);
      v16bf b = load_frag16(Wt1 + (nt * 16 + coll) * E::K2P + kt * 32 + kb);
      acc = wmma_bf16(a, b, acc);
    }
    float s = 0.f, q = 0.f;
#pragma unroll
    for (int r = 0; r < 8; ++r) { float v = acc[r]; s += v; q += v * v; }
    atomicAdd(&cs[nt * 16 + coll], s);
    atomicAdd(&cq[nt * 16 + coll], q);
  }
  __syncthreads();
  for (int t = tid; t < COUT; t += 256) {
    atomicAdd(&gsum[t], cs[t]);
    atomicAdd(&gsq[t], cq[t]);
  }
}

// Pass 2: recompute z1, BN1+ReLU -> h1 (LDS bf16), z2 = h1 @ W2, stats of z2.
template <int CIN, int COUT, int MT>
__global__ __launch_bounds__(256) void edge_gemm2_stats(
    const float* __restrict__ x, const int* __restrict__ idxb,
    const unsigned short* __restrict__ Wt1, const unsigned short* __restrict__ Wt2,
    const float* __restrict__ s1, const float* __restrict__ q1,
    const float* __restrict__ g1, const float* __restrict__ bt1,
    float* __restrict__ gsum2, float* __restrict__ gsq2) {
  using E = EC<CIN, COUT, MT>;
  extern __shared__ char smem[];
  unsigned short* As = (unsigned short*)smem;
  unsigned short* Hs = (unsigned short*)(smem + (size_t)MT * E::K2P * 2);
  float* ssc = (float*)(smem + (size_t)MT * E::K2P * 2 + (size_t)MT * COUT * 2);
  float* ssh = ssc + COUT;
  float* cs = ssh + COUT;
  float* cq = cs + COUT;
  int* jrow = (int*)(cq + COUT);
  const int tid = threadIdx.x;
  const int e0 = blockIdx.x * MT;
  constexpr float INVM = 1.f / 1048576.f;

  for (int t = tid; t < COUT; t += 256) {
    float m = s1[t] * INVM;
    float var = q1[t] * INVM - m * m;
    float sc = g1[t] * rsqrtf(var + EPSV);
    ssc[t] = sc;
    ssh[t] = bt1[t] - m * sc;
    cs[t] = 0.f;
    cq[t] = 0.f;
  }
  build_Atile<CIN, COUT, MT>(x, idxb, As, jrow, e0, tid);

  const int wid = tid >> 5, lane = tid & 31;
  const int rg = wid % E::RG, nt0 = wid / E::RG;
  const int rowb = rg * 16 + (lane & 15);
  const int kb = (lane >> 4) * 8;
  const int coll = lane & 15;
  const int rhalf = 8 * (lane >> 4);

  // GEMM1 -> Hs (BN1 + ReLU, bf16)
  for (int ni = 0; ni < E::NTW; ++ni) {
    const int nt = nt0 + ni * E::NS;
    v8f acc = {0.f, 0.f, 0.f, 0.f, 0.f, 0.f, 0.f, 0.f};
#pragma unroll
    for (int kt = 0; kt < E::K2P / 32; ++kt) {
      v16bf a = load_frag16(As + rowb * E::K2P + kt * 32 + kb);
      v16bf b = load_frag16(Wt1 + (nt * 16 + coll) * E::K2P + kt * 32 + kb);
      acc = wmma_bf16(a, b, acc);
    }
    const int col = nt * 16 + coll;
    const float sc = ssc[col], sh = ssh[col];
#pragma unroll
    for (int r = 0; r < 8; ++r) {
      int row = rg * 16 + r + rhalf;
      Hs[row * COUT + col] = f2bf(fmaxf(sc * acc[r] + sh, 0.f));
    }
  }
  __syncthreads();

  // GEMM2 -> column stats of z2
  for (int ni = 0; ni < E::NTW; ++ni) {
    const int nt = nt0 + ni * E::NS;
    v8f acc = {0.f, 0.f, 0.f, 0.f, 0.f, 0.f, 0.f, 0.f};
#pragma unroll
    for (int kt = 0; kt < COUT / 32; ++kt) {
      v16bf a = load_frag16(Hs + rowb * COUT + kt * 32 + kb);
      v16bf b = load_frag16(Wt2 + (nt * 16 + coll) * COUT + kt * 32 + kb);
      acc = wmma_bf16(a, b, acc);
    }
    float s = 0.f, q = 0.f;
#pragma unroll
    for (int r = 0; r < 8; ++r) { float v = acc[r]; s += v; q += v * v; }
    atomicAdd(&cs[nt * 16 + coll], s);
    atomicAdd(&cq[nt * 16 + coll], q);
  }
  __syncthreads();
  for (int t = tid; t < COUT; t += 256) {
    atomicAdd(&gsum2[t], cs[t]);
    atomicAdd(&gsq2[t], cq[t]);
  }
}

// Pass 3: recompute z1->h1->z2, BN2+ReLU, mean over the 16 edges -> x_out[N,COUT]
template <int CIN, int COUT, int MT>
__global__ __launch_bounds__(256) void edge_final(
    const float* __restrict__ x, const int* __restrict__ idxb,
    const unsigned short* __restrict__ Wt1, const unsigned short* __restrict__ Wt2,
    const float* __restrict__ s1, const float* __restrict__ q1,
    const float* __restrict__ g1, const float* __restrict__ bt1,
    const float* __restrict__ s2, const float* __restrict__ q2,
    const float* __restrict__ g2, const float* __restrict__ bt2,
    float* __restrict__ xout) {
  using E = EC<CIN, COUT, MT>;
  extern __shared__ char smem[];
  unsigned short* As = (unsigned short*)smem;
  unsigned short* Hs = (unsigned short*)(smem + (size_t)MT * E::K2P * 2);
  float* ssc = (float*)(smem + (size_t)MT * E::K2P * 2 + (size_t)MT * COUT * 2);
  float* ssh = ssc + COUT;
  float* sc2 = ssh + COUT;
  float* sh2 = sc2 + COUT;
  int* jrow = (int*)(sh2 + COUT);
  const int tid = threadIdx.x;
  const int e0 = blockIdx.x * MT;
  constexpr float INVM = 1.f / 1048576.f;

  for (int t = tid; t < COUT; t += 256) {
    float m = s1[t] * INVM;
    float var = q1[t] * INVM - m * m;
    float sc = g1[t] * rsqrtf(var + EPSV);
    ssc[t] = sc;
    ssh[t] = bt1[t] - m * sc;
    float m2 = s2[t] * INVM;
    float var2 = q2[t] * INVM - m2 * m2;
    float c2 = g2[t] * rsqrtf(var2 + EPSV);
    sc2[t] = c2;
    sh2[t] = bt2[t] - m2 * c2;
  }
  build_Atile<CIN, COUT, MT>(x, idxb, As, jrow, e0, tid);

  const int wid = tid >> 5, lane = tid & 31;
  const int rg = wid % E::RG, nt0 = wid / E::RG;
  const int rowb = rg * 16 + (lane & 15);
  const int kb = (lane >> 4) * 8;
  const int coll = lane & 15;
  const int rhalf = 8 * (lane >> 4);

  for (int ni = 0; ni < E::NTW; ++ni) {
    const int nt = nt0 + ni * E::NS;
    v8f acc = {0.f, 0.f, 0.f, 0.f, 0.f, 0.f, 0.f, 0.f};
#pragma unroll
    for (int kt = 0; kt < E::K2P / 32; ++kt) {
      v16bf a = load_frag16(As + rowb * E::K2P + kt * 32 + kb);
      v16bf b = load_frag16(Wt1 + (nt * 16 + coll) * E::K2P + kt * 32 + kb);
      acc = wmma_bf16(a, b, acc);
    }
    const int col = nt * 16 + coll;
    const float sc = ssc[col], sh = ssh[col];
#pragma unroll
    for (int r = 0; r < 8; ++r) {
      int row = rg * 16 + r + rhalf;
      Hs[row * COUT + col] = f2bf(fmaxf(sc * acc[r] + sh, 0.f));
    }
  }
  __syncthreads();

  // GEMM2 + BN2/ReLU + mean over the wave's 16 edge rows (= one node)
  for (int ni = 0; ni < E::NTW; ++ni) {
    const int nt = nt0 + ni * E::NS;
    v8f acc = {0.f, 0.f, 0.f, 0.f, 0.f, 0.f, 0.f, 0.f};
#pragma unroll
    for (int kt = 0; kt < COUT / 32; ++kt) {
      v16bf a = load_frag16(Hs + rowb * COUT + kt * 32 + kb);
      v16bf b = load_frag16(Wt2 + (nt * 16 + coll) * COUT + kt * 32 + kb);
      acc = wmma_bf16(a, b, acc);
    }
    const int col = nt * 16 + coll;
    const float sc = sc2[col], sh = sh2[col];
    float s = 0.f;
#pragma unroll
    for (int r = 0; r < 8; ++r) s += fmaxf(sc * acc[r] + sh, 0.f);
    s += __shfl_xor(s, 16, 32);  // combine the two row-halves of the C tile
    if (lane < 16) {
      int node = blockIdx.x * E::RG + rg;
      xout[node * COUT + col] = s * (1.f / 16.f);
    }
  }
}

// ---------------------------------------------------------------------------
// Host-side per-block driver
// ---------------------------------------------------------------------------
template <int CIN, int COUT, int MT>
static void run_block(const float* xcur, float* xnext, int* idxb,
                      unsigned short* wt1, unsigned short* wt2,
                      const float* W1, const float* g1, const float* bt1,
                      const float* W2, const float* g2, const float* bt2,
                      float* s1, float* s2, float* pooled, int off,
                      hipStream_t stream) {
  using E = EC<CIN, COUT, MT>;
  constexpr int GRID = (512 * 128 * 16) / MT;
  float* q1 = s1 + 256;
  float* q2 = s2 + 256;
  int nW1 = COUT * E::K2P;
  convw_kernel<<<(nW1 + 255) / 256, 256, 0, stream>>>(W1, wt1, E::K2, COUT, E::K2P);
  int nW2 = COUT * COUT;
  convw_kernel<<<(nW2 + 255) / 256, 256, 0, stream>>>(W2, wt2, COUT, COUT, COUT);
  knn_kernel<CIN><<<512, 128, 0, stream>>>(xcur, idxb);
  zero_kernel<<<2, 256, 0, stream>>>(s1, 512);
  edge_gemm1_stats<CIN, COUT, MT><<<GRID, 256, E::SM1, stream>>>(
      xcur, idxb, wt1, s1, q1);
  zero_kernel<<<2, 256, 0, stream>>>(s2, 512);
  edge_gemm2_stats<CIN, COUT, MT><<<GRID, 256, E::SM23, stream>>>(
      xcur, idxb, wt1, wt2, s1, q1, g1, bt1, s2, q2);
  edge_final<CIN, COUT, MT><<<GRID, 256, E::SM23, stream>>>(
      xcur, idxb, wt1, wt2, s1, q1, g1, bt1, s2, q2, g2, bt2, xnext);
  pool_kernel<COUT><<<512, 256, 0, stream>>>(xnext, pooled, off);
}

extern "C" void kernel_launch(void* const* d_in, const int* in_sizes, int n_in,
                              void* d_out, int out_size, void* d_ws, size_t ws_size,
                              hipStream_t stream) {
  (void)in_sizes; (void)n_in; (void)out_size; (void)ws_size;
  const int B = 512, P = 128, N = B * P;

  // Input order: recursive insertion order of setup_inputs() dict.
  const float* x    = (const float*)d_in[0];
  const float* in_g = (const float*)d_in[3];
  const float* in_b = (const float*)d_in[4];
  // blocks: W1,b1,g1,bt1,W2,b2,g2,bt2  (b1/b2 unused: cancelled by BN)
  const float* bW1[3] = {(const float*)d_in[5],  (const float*)d_in[13], (const float*)d_in[21]};
  const float* bG1[3] = {(const float*)d_in[7],  (const float*)d_in[15], (const float*)d_in[23]};
  const float* bT1[3] = {(const float*)d_in[8],  (const float*)d_in[16], (const float*)d_in[24]};
  const float* bW2[3] = {(const float*)d_in[9],  (const float*)d_in[17], (const float*)d_in[25]};
  const float* bG2[3] = {(const float*)d_in[11], (const float*)d_in[19], (const float*)d_in[27]};
  const float* bT2[3] = {(const float*)d_in[12], (const float*)d_in[20], (const float*)d_in[28]};
  const float* fW0 = (const float*)d_in[29];
  const float* fG0 = (const float*)d_in[31];
  const float* fT0 = (const float*)d_in[32];
  const float* fW1 = (const float*)d_in[33];
  const float* fG1 = (const float*)d_in[35];
  const float* fT1 = (const float*)d_in[36];
  const float* oW  = (const float*)d_in[37];
  const float* oB  = (const float*)d_in[38];

  // Workspace layout (~140 MB)
  char* ws = (char*)d_ws;
  float* xa = (float*)ws;                                       // N*256 f32
  float* xb = (float*)(ws + (size_t)67108864);                  // N*256 f32
  int* idxb = (int*)(ws + (size_t)134217728);                   // N*16 int
  unsigned short* wt1 = (unsigned short*)(ws + (size_t)138412032);  // 128KB
  unsigned short* wt2 = (unsigned short*)(ws + (size_t)138543104);  // 128KB
  float* stats = (float*)(ws + (size_t)138674176);              // 2048 f32
  float* pooled = (float*)(ws + (size_t)138682368);             // 512*448
  float* fb1 = (float*)(ws + (size_t)139599872);                // 512*256
  float* fb2 = (float*)(ws + (size_t)140124160);                // 512*128

  float* sIn = stats;          // +qIn at +256
  float* s1  = stats + 512;    // +q1 at +768
  float* s2  = stats + 1024;   // +q2 at +1280
  float* sf  = stats + 1536;   // +qf at +1792

  // ---- input BatchNorm ----
  zero_kernel<<<2, 256, 0, stream>>>(sIn, 512);
  colstats_kernel<<<64, 256, 2 * 4 * sizeof(float), stream>>>(x, N, 4, sIn, sIn + 256);
  bn_input_kernel<<<(N * 4 + 255) / 256, 256, 0, stream>>>(
      x, sIn, sIn + 256, in_g, in_b, xa, N, 4);

  // ---- EdgeConv blocks ----
  run_block<4, 64, 64>(xa, xb, idxb, wt1, wt2, bW1[0], bG1[0], bT1[0],
                       bW2[0], bG2[0], bT2[0], s1, s2, pooled, 0, stream);
  run_block<64, 128, 64>(xb, xa, idxb, wt1, wt2, bW1[1], bG1[1], bT1[1],
                         bW2[1], bG2[1], bT2[1], s1, s2, pooled, 64, stream);
  run_block<128, 256, 32>(xa, xb, idxb, wt1, wt2, bW1[2], bG1[2], bT1[2],
                          bW2[2], bG2[2], bT2[2], s1, s2, pooled, 192, stream);

  // ---- FC tail ----
  fc_gemm_kernel<<<(512 * 256 + 255) / 256, 256, 0, stream>>>(pooled, fW0, fb1, 512, 448, 256);
  zero_kernel<<<2, 256, 0, stream>>>(sf, 512);
  colstats_kernel<<<8, 256, 2 * 256 * sizeof(float), stream>>>(fb1, 512, 256, sf, sf + 256);
  bn_relu_kernel<<<(512 * 256 + 255) / 256, 256, 0, stream>>>(fb1, sf, sf + 256, fG0, fT0, 512, 256);

  fc_gemm_kernel<<<(512 * 128 + 255) / 256, 256, 0, stream>>>(fb1, fW1, fb2, 512, 256, 128);
  zero_kernel<<<2, 256, 0, stream>>>(sf, 512);
  colstats_kernel<<<8, 256, 2 * 128 * sizeof(float), stream>>>(fb2, 512, 128, sf, sf + 256);
  bn_relu_kernel<<<(512 * 128 + 255) / 256, 256, 0, stream>>>(fb2, sf, sf + 256, fG1, fT1, 512, 128);

  out_kernel<<<(512 * 2 + 255) / 256, 256, 0, stream>>>(fb2, oW, oB, (float*)d_out, 512, 128, 2);
}
